// SAHead_30631706755552
// MI455X (gfx1250) — matrix operand order repaired and checked by reading, not compiled, not run on hardware
//
#include <hip/hip_runtime.h>
#include <stdint.h>

// ---------------- types for WMMA fragments ----------------
typedef float  v8f   __attribute__((ext_vector_type(8)));
typedef __bf16 v16bf __attribute__((ext_vector_type(16)));

union Frag {
    v16bf    h;
    uint32_t u[8];
};

__device__ __forceinline__ uint32_t bf16rne_bits(float a) {
    uint32_t ua = __float_as_uint(a);
    ua = (ua + 0x7FFFu + ((ua >> 16) & 1u)) >> 16;
    return ua;
}
__device__ __forceinline__ uint32_t pk_bf16(float a, float b) {
    return bf16rne_bits(a) | (bf16rne_bits(b) << 16);
}

#define L_TOT 4096
#define C_IN  320
#define CE_IN 326
#define DK    64
#define JSTEP 64       // keys staged per LDS stage
#define ROWP  72       // padded LDS row stride in bf16 elements (144B, spreads banks)

// ---------------- CDNA5 async global->LDS path ----------------
#if defined(__has_builtin)
#  if __has_builtin(__builtin_amdgcn_global_load_async_to_lds_b128)
#    define USE_ASYNC 1
#  endif
#endif
#ifndef USE_ASYNC
#  define USE_ASYNC 0
#endif

// Builtin prototype (from hipcc diagnostic):
//   param0: int4(vector_size 16) pointer, AS1 (global), non-const
//   param1: int4(vector_size 16) pointer, AS3 (LDS)
//   param2/3: constant ints (offset, cpol)
typedef int v4i_vs __attribute__((vector_size(16)));
typedef __attribute__((address_space(1))) v4i_vs* gbl128_t;
typedef __attribute__((address_space(3))) v4i_vs* lds128_t;

__device__ __forceinline__ void cp16_async(uint16_t* lptr, const uint16_t* gptr) {
#if USE_ASYNC
    __builtin_amdgcn_global_load_async_to_lds_b128(
        (gbl128_t)(uintptr_t)gptr,
        (lds128_t)(uint32_t)(uintptr_t)lptr,
        0, 0);
#else
    *(uint4*)lptr = *(const uint4*)gptr;
#endif
}

#if USE_ASYNC
#  if __has_builtin(__builtin_amdgcn_s_wait_asynccnt)
#    define WAIT_ASYNC(N) __builtin_amdgcn_s_wait_asynccnt(N)
#  else
#    define WAIT_ASYNC(N) asm volatile("s_wait_asynccnt %0" :: "i"(N) : "memory")
#  endif
#else
#  define WAIT_ASYNC(N) ((void)0)
#endif

// ============================================================
// Kernel 1: fused QKV projection + L2 normalize (channel axis)
//   Qn, Kn: bf16 [B, L, 64]  (position-major -> contiguous rows)
//   Vt    : bf16 [B, 64, L]  (channel-major  -> contiguous rows)
// grid = B * L/4 blocks, 256 threads (64 channels x 4 positions)
// ============================================================
__global__ __launch_bounds__(256) void proj_kernel(
    const float* __restrict__ x, const float* __restrict__ xe,
    const float* __restrict__ Wq, const float* __restrict__ bq,
    const float* __restrict__ Wk, const float* __restrict__ bk,
    const float* __restrict__ Wv, const float* __restrict__ bv,
    uint16_t* __restrict__ Qn, uint16_t* __restrict__ Kn,
    uint16_t* __restrict__ Vt)
{
    __shared__ float sxe[CE_IN * 4];
    __shared__ float sx [C_IN * 4];
    __shared__ float red[2][256];
    __shared__ float nrm[2][4];

    const int blk = blockIdx.x;
    const int b   = blk >> 10;            // L/4 = 1024 blocks per batch
    const int l0  = (blk & 1023) << 2;
    const int t   = threadIdx.x;
    const int o   = t >> 2;               // channel 0..63
    const int p   = t & 3;                // position within tile

    const float* xeb = xe + (size_t)b * CE_IN * L_TOT;
    const float* xb  = x  + (size_t)b * C_IN  * L_TOT;

    for (int i = t; i < CE_IN * 4; i += 256) {
        int c = i >> 2, pp = i & 3;
        sxe[i] = xeb[(size_t)c * L_TOT + l0 + pp];
    }
    for (int i = t; i < C_IN * 4; i += 256) {
        int c = i >> 2, pp = i & 3;
        sx[i] = xb[(size_t)c * L_TOT + l0 + pp];
    }
    __syncthreads();

    float q = bq[o], k = bk[o], v = bv[o];
    const float* wq = Wq + (size_t)o * CE_IN;
    const float* wk = Wk + (size_t)o * CE_IN;
    const float* wv = Wv + (size_t)o * C_IN;
    for (int c = 0; c < CE_IN; ++c) {
        float xv = sxe[c * 4 + p];
        q = fmaf(wq[c], xv, q);
        k = fmaf(wk[c], xv, k);
    }
    for (int c = 0; c < C_IN; ++c)
        v = fmaf(wv[c], sx[c * 4 + p], v);

    red[0][p * 64 + o] = q * q;
    red[1][p * 64 + o] = k * k;
    __syncthreads();
    if (t < 8) {
        int which = t >> 2, pp = t & 3;
        float s = 0.f;
        for (int i = 0; i < 64; ++i) s += red[which][pp * 64 + i];
        nrm[which][pp] = fmaxf(sqrtf(s), 1e-6f);
    }
    __syncthreads();

    float qn = q / nrm[0][p];
    float kn = k / nrm[1][p];

    size_t pos = ((size_t)(b * L_TOT + l0 + p)) * DK + o;
    Qn[pos] = (uint16_t)bf16rne_bits(qn);
    Kn[pos] = (uint16_t)bf16rne_bits(kn);
    Vt[((size_t)(b * DK + o)) * L_TOT + l0 + p] = (uint16_t)bf16rne_bits(v);
}

// ============================================================
// Kernel 2: flash attention.
//   Workgroup = 4 waves (128 thr), 64 queries (16/wave).
//   K/V staged to double-buffered LDS via async global->LDS,
//   shared by all 4 waves (4x L2 traffic reduction).
//   Per 32-key substep: S^T = Kn x Qn^T (2 WMMA k-steps),
//   online softmax, O^T += V x P^T (4 dv tiles).
// grid = B * L/64 = 128 blocks, 128 threads.
// ============================================================
__device__ __forceinline__ void stage_load(uint16_t* sK, uint16_t* sV,
                                           const uint16_t* Kb, const uint16_t* Vb,
                                           int j0, int t)
{
    // K tile: 64 keys x 64 ch; V tile: 64 ch x 64 keys. 512 16B-chunks each.
#pragma unroll
    for (int i = 0; i < 4; ++i) {
        int chunk = t + i * 128;            // 0..511
        int row = chunk >> 3, c8 = chunk & 7;
        cp16_async(sK + row * ROWP + c8 * 8,
                   Kb + (size_t)(j0 + row) * DK + c8 * 8);
    }
#pragma unroll
    for (int i = 0; i < 4; ++i) {
        int chunk = t + i * 128;
        int row = chunk >> 3, c8 = chunk & 7;
        cp16_async(sV + row * ROWP + c8 * 8,
                   Vb + (size_t)row * L_TOT + j0 + c8 * 8);
    }
}

__global__ __launch_bounds__(128) void attn_kernel(
    const uint16_t* __restrict__ Qn, const uint16_t* __restrict__ Kn,
    const uint16_t* __restrict__ Vt, float* __restrict__ out)
{
    __shared__ uint16_t sK[2][JSTEP * ROWP];
    __shared__ uint16_t sV[2][DK * ROWP];

    const int b    = blockIdx.x >> 6;          // 64 q-blocks per batch
    const int t    = threadIdx.x;
    const int wave = t >> 5;
    const int lane = t & 31;
    const int n    = lane & 15;
    const int hi   = lane >> 4;
    const int q0   = ((blockIdx.x & 63) << 6) + (wave << 4);

    const uint16_t* Qb = Qn + (size_t)b * L_TOT * DK;
    const uint16_t* Kb = Kn + (size_t)b * L_TOT * DK;
    const uint16_t* Vb = Vt + (size_t)b * DK * L_TOT;

    // B-fragments of Qn^T: lane n holds column (query q0+n), K = channels.
    Frag bq0, bq1;
    {
        const uint32_t* p0 = (const uint32_t*)(Qb + (size_t)(q0 + n) * DK);
        const uint32_t* p1 = (const uint32_t*)(Qb + (size_t)(q0 + n) * DK + 32);
#pragma unroll
        for (int r = 0; r < 4; ++r) { bq0.u[r] = p0[hi * 4 + r]; bq0.u[4 + r] = p0[8 + hi * 4 + r]; bq1.u[r] = p1[hi * 4 + r]; bq1.u[4 + r] = p1[8 + hi * 4 + r]; }
    }

    float m = -1e30f, lsum = 0.f;
    v8f O0 = {}, O1 = {}, O2 = {}, O3 = {};

    stage_load(sK[0], sV[0], Kb, Vb, 0, t);

    const int NSTAGE = L_TOT / JSTEP;          // 64
    for (int s = 0; s < NSTAGE; ++s) {
        const int cur = s & 1;
        const int j0n = (s + 1) * JSTEP;
        if (j0n < L_TOT) {
            stage_load(sK[cur ^ 1], sV[cur ^ 1], Kb, Vb, j0n, t);
            WAIT_ASYNC(8);                     // retire current stage, next in flight
        } else {
            WAIT_ASYNC(0);
        }
        __syncthreads();                       // all waves' stage data visible

        const uint16_t* sKb = sK[cur];
        const uint16_t* sVb = sV[cur];

#pragma unroll
        for (int js = 0; js < JSTEP; js += 32) {
            v8f d0 = {}, d1 = {};
            // --- S^T tile 0: local keys js..js+15 ---
            {
                const uint32_t* pk0 = (const uint32_t*)(sKb + (js + n) * ROWP);
                Frag a0, a1;
#pragma unroll
                for (int r = 0; r < 4; ++r) {
                    a0.u[r]     = pk0[hi * 4 + r];
                    a0.u[4 + r] = pk0[8 + hi * 4 + r];
                    a1.u[r]     = pk0[16 + hi * 4 + r];
                    a1.u[4 + r] = pk0[24 + hi * 4 + r];
                }
                d0 = __builtin_amdgcn_wmma_f32_16x16x32_bf16(false, a0.h, false, bq0.h, (short)0, d0, false, false);
                d0 = __builtin_amdgcn_wmma_f32_16x16x32_bf16(false, a1.h, false, bq1.h, (short)0, d0, false, false);
            }
            // --- S^T tile 1: local keys js+16..js+31 ---
            {
                const uint32_t* pk1 = (const uint32_t*)(sKb + (js + 16 + n) * ROWP);
                Frag a0, a1;
#pragma unroll
                for (int r = 0; r < 4; ++r) {
                    a0.u[r]     = pk1[hi * 4 + r];
                    a0.u[4 + r] = pk1[8 + hi * 4 + r];
                    a1.u[r]     = pk1[16 + hi * 4 + r];
                    a1.u[4 + r] = pk1[24 + hi * 4 + r];
                }
                d1 = __builtin_amdgcn_wmma_f32_16x16x32_bf16(false, a0.h, false, bq0.h, (short)0, d1, false, false);
                d1 = __builtin_amdgcn_wmma_f32_16x16x32_bf16(false, a1.h, false, bq1.h, (short)0, d1, false, false);
            }

            // --- online softmax over the 32 keys (query column i = lane&15) ---
            float mx = d0[0];
#pragma unroll
            for (int r = 1; r < 8; ++r) mx = fmaxf(mx, d0[r]);
#pragma unroll
            for (int r = 0; r < 8; ++r) mx = fmaxf(mx, d1[r]);
            mx = fmaxf(mx, __shfl_xor(mx, 16, 32));
            float m_new = fmaxf(m, mx);
            float alpha = __expf(m - m_new);

            float p0v[8], p1v[8];
            float ls = 0.f;
#pragma unroll
            for (int r = 0; r < 8; ++r) { p0v[r] = __expf(d0[r] - m_new); ls += p0v[r]; }
#pragma unroll
            for (int r = 0; r < 8; ++r) { p1v[r] = __expf(d1[r] - m_new); ls += p1v[r]; }
            ls += __shfl_xor(ls, 16, 32);
            lsum = lsum * alpha + ls;
            m = m_new;

#pragma unroll
            for (int r = 0; r < 8; ++r) {
                O0[r] *= alpha; O1[r] *= alpha; O2[r] *= alpha; O3[r] *= alpha;
            }

            // --- build P^T B-fragment (32j x 16i), bf16, half-wave swap ---
            uint32_t t1p[4], t2p[4];
#pragma unroll
            for (int r = 0; r < 4; ++r) {
                t1p[r] = pk_bf16(p0v[2 * r], p0v[2 * r + 1]);
                t2p[r] = pk_bf16(p1v[2 * r], p1v[2 * r + 1]);
            }
            Frag bp;
#pragma unroll
            for (int r = 0; r < 4; ++r) {
                uint32_t o1 = (uint32_t)__shfl_xor((int)t1p[r], 16, 32);
                uint32_t o2 = (uint32_t)__shfl_xor((int)t2p[r], 16, 32);
                bp.u[r]     = hi ? o2     : t1p[r];
                bp.u[4 + r] = hi ? t2p[r] : o1;
            }

            // --- O^T += V * P^T, 4 tiles over dv ---
#pragma unroll
            for (int vt = 0; vt < 4; ++vt) {
                const uint32_t* pv = (const uint32_t*)(sVb + (vt * 16 + n) * ROWP + js);
                Frag av;
#pragma unroll
                for (int r = 0; r < 4; ++r) {
                    av.u[r]     = pv[hi * 8 + r];
                    av.u[4 + r] = pv[hi * 8 + 4 + r];
                }
                if      (vt == 0) O0 = __builtin_amdgcn_wmma_f32_16x16x32_bf16(false, av.h, false, bp.h, (short)0, O0, false, false);
                else if (vt == 1) O1 = __builtin_amdgcn_wmma_f32_16x16x32_bf16(false, av.h, false, bp.h, (short)0, O1, false, false);
                else if (vt == 2) O2 = __builtin_amdgcn_wmma_f32_16x16x32_bf16(false, av.h, false, bp.h, (short)0, O2, false, false);
                else              O3 = __builtin_amdgcn_wmma_f32_16x16x32_bf16(false, av.h, false, bp.h, (short)0, O3, false, false);
            }
        }
        __syncthreads();                       // done reading buf before it is refilled
    }

    // --- finalize: R[b, v, i] = O^T[v, i] / l_i ---
    float inv = 1.f / lsum;
    float* ob = out + (size_t)b * DK * L_TOT;
#pragma unroll
    for (int r = 0; r < 8; ++r) {
        int row = hi * 8 + r;
        ob[(size_t)(row)      * L_TOT + q0 + n] = O0[r] * inv;
        ob[(size_t)(row + 16) * L_TOT + q0 + n] = O1[r] * inv;
        ob[(size_t)(row + 32) * L_TOT + q0 + n] = O2[r] * inv;
        ob[(size_t)(row + 48) * L_TOT + q0 + n] = O3[r] * inv;
    }
}

// ============================================================
extern "C" void kernel_launch(void* const* d_in, const int* in_sizes, int n_in,
                              void* d_out, int out_size, void* d_ws, size_t ws_size,
                              hipStream_t stream)
{
    (void)in_sizes; (void)n_in; (void)out_size; (void)ws_size;
    const float* x   = (const float*)d_in[0];
    const float* xe  = (const float*)d_in[1];
    const float* Wq  = (const float*)d_in[2];
    const float* bq  = (const float*)d_in[3];
    const float* Wk  = (const float*)d_in[4];
    const float* bk  = (const float*)d_in[5];
    const float* Wv  = (const float*)d_in[6];
    const float* bv  = (const float*)d_in[7];

    uint16_t* Qn = (uint16_t*)d_ws;                       // [2, 4096, 64] bf16
    uint16_t* Kn = Qn + (size_t)2 * L_TOT * DK;           // [2, 4096, 64] bf16
    uint16_t* Vt = Kn + (size_t)2 * L_TOT * DK;           // [2, 64, 4096] bf16

    proj_kernel<<<2048, 256, 0, stream>>>(x, xe, Wq, bq, Wk, bk, Wv, bv, Qn, Kn, Vt);
    attn_kernel<<<128, 128, 0, stream>>>(Qn, Kn, Vt, (float*)d_out);
}
